// GLVQ_86114094284878
// MI455X (gfx1250) — matrix operand order repaired and checked by reading, not compiled, not run on hardware
//
#include <hip/hip_runtime.h>
#include <math.h>

#define BATCH 16384
#define FEAT  1024
#define NALL  1000
#define NPAD  1024
#define NCLS  100
#define NPROT 10

typedef __attribute__((ext_vector_type(16))) __bf16 bf16x16;
typedef __attribute__((ext_vector_type(8)))  __bf16 bf16x8;
typedef __attribute__((ext_vector_type(8)))  float  f32x8;

union Frag { bf16x16 v; bf16x8 h[2]; };

// ---------- async global->LDS copy path (CDNA5), with safe fallback ----------
#if defined(__has_builtin)
#if __has_builtin(__builtin_amdgcn_global_load_async_to_lds_b128)
#define ASYNC_OK 1
#endif
#endif

#if defined(ASYNC_OK)
#if __has_builtin(__builtin_amdgcn_s_wait_asynccnt)
#define WAIT_ASYNC(n) __builtin_amdgcn_s_wait_asynccnt(n)
#else
#define WAIT_ASYNC(n) asm volatile("s_wait_asynccnt %0" ::"i"(n) : "memory")
#endif
#else
#define WAIT_ASYNC(n)
#endif

// Types matching the builtin's declared parameters:
//   arg0: v4i32 in global addrspace(1), arg1: v4i32 in LDS addrspace(3)
typedef int i32x4v __attribute__((vector_size(16)));
typedef __attribute__((address_space(1))) i32x4v* gptr_b128;
typedef __attribute__((address_space(3))) i32x4v* lptr_b128;

__device__ __forceinline__ void async_copy_b128(const unsigned short* g, unsigned short* l) {
#if defined(ASYNC_OK)
    __builtin_amdgcn_global_load_async_to_lds_b128((gptr_b128)g, (lptr_b128)l, 0, 0);
#else
    *(uint4*)l = *(const uint4*)g;
#endif
}

// ---------- bf16 split helpers (RNE, bit-level) ----------
__device__ __forceinline__ unsigned short f32_to_bf16(float f) {
    unsigned int u = __float_as_uint(f);
    u += 0x7FFFu + ((u >> 16) & 1u);
    return (unsigned short)(u >> 16);
}
__device__ __forceinline__ float bf16f(unsigned short h) {
    return __uint_as_float(((unsigned int)h) << 16);
}
__device__ __forceinline__ void split2(float v, unsigned short& hi, unsigned short& lo) {
    hi = f32_to_bf16(v);
    lo = f32_to_bf16(v - bf16f(hi));
}
__device__ __forceinline__ void split_store4(float4 v, unsigned short* hp, unsigned short* lp) {
    unsigned short h0,h1,h2,h3,l0,l1,l2,l3;
    split2(v.x,h0,l0); split2(v.y,h1,l1); split2(v.z,h2,l2); split2(v.w,h3,l3);
    uint2 H, L;
    H.x = (unsigned)h0 | ((unsigned)h1 << 16); H.y = (unsigned)h2 | ((unsigned)h3 << 16);
    L.x = (unsigned)l0 | ((unsigned)l1 << 16); L.y = (unsigned)l2 | ((unsigned)l3 << 16);
    *(uint2*)hp = H; *(uint2*)lp = L;
}

// ---------- WMMA ----------
__device__ __forceinline__ f32x8 wmma_bf16(const Frag& a, const Frag& b, f32x8 c) {
    return __builtin_amdgcn_wmma_f32_16x16x32_bf16(false, a.v, false, b.v,
                                                   (short)0, c, false, false);
}

// A-operand 16x32 frag from LDS [row][stride 40]:
//  lane<16: rows 0..15, K = {0..7, 16..23}; lane>=16: K = {8..15, 24..31}
__device__ __forceinline__ void load_frag_a(const unsigned short* base, int row, int kh, Frag& f) {
    const unsigned short* p = base + row * 40 + kh * 8;
    f.h[0] = *(const bf16x8*)p;
    f.h[1] = *(const bf16x8*)(p + 16);
}
// B-operand 32x16 frag from LDS [col][stride 40]:
//  lane<16: cols 0..15, K = {0..7, 16..23}; lane>=16: K = {8..15, 24..31}
__device__ __forceinline__ void load_frag_b(const unsigned short* base, int col, int kh, Frag& f) {
    const unsigned short* p = base + col * 40 + kh * 8;
    f.h[0] = *(const bf16x8*)p;
    f.h[1] = *(const bf16x8*)(p + 16);
}

// ---------- kernel: split x into bf16 hi/lo + row sum of squares (fused single pass) ----------
__global__ void __launch_bounds__(256) xsplit_kernel(const float* __restrict__ x,
                                                     unsigned short* __restrict__ xhi,
                                                     unsigned short* __restrict__ xlo,
                                                     float* __restrict__ xsq) {
    const int wid = threadIdx.x >> 5, lane = threadIdx.x & 31;
    const int row = blockIdx.x * 8 + wid;
    const size_t base = (size_t)row * FEAT;
    float s = 0.f;
    #pragma unroll
    for (int i = 0; i < FEAT / 128; ++i) {
        const int col = i * 128 + lane * 4;
        float4 v = *(const float4*)&x[base + col];
        if (i + 1 < FEAT / 128) __builtin_prefetch(&x[base + col + 128], 0, 0);
        s += v.x * v.x + v.y * v.y + v.z * v.z + v.w * v.w;
        split_store4(v, &xhi[base + col], &xlo[base + col]);
    }
    #pragma unroll
    for (int off = 16; off; off >>= 1) s += __shfl_xor(s, off, 32);
    if (lane == 0) xsq[row] = s;
}

// ---------- kernel: p_sq from split ep ----------
__global__ void __launch_bounds__(256) psq_kernel(const unsigned short* __restrict__ hi,
                                                  const unsigned short* __restrict__ lo,
                                                  float* __restrict__ out) {
    const int wid = threadIdx.x >> 5, lane = threadIdx.x & 31;
    const int row = blockIdx.x * 8 + wid;
    const size_t base = (size_t)row * FEAT;
    float s = 0.f;
    for (int k = lane; k < FEAT; k += 32) {
        float v = bf16f(hi[base + k]) + bf16f(lo[base + k]);
        s += v * v;
    }
    #pragma unroll
    for (int off = 16; off; off >>= 1) s += __shfl_xor(s, off, 32);
    if (lane == 0) out[row] = s;
}

// ---------- kernel: encoder GEMM  ep = protos @ enc_w + enc_b (split to bf16 hi/lo) ----------
// block: 256 thr (8 waves), tile 128(j) x 64(f), K step 32
__global__ void __launch_bounds__(256) enc_gemm_kernel(const float* __restrict__ protos,
                                                       const float* __restrict__ encw,
                                                       const float* __restrict__ encb,
                                                       unsigned short* __restrict__ ep_hi,
                                                       unsigned short* __restrict__ ep_lo) {
    __shared__ __align__(16) unsigned short sAhi[128 * 40];
    __shared__ __align__(16) unsigned short sAlo[128 * 40];
    __shared__ __align__(16) unsigned short sBhi[64 * 40];
    __shared__ __align__(16) unsigned short sBlo[64 * 40];

    const int tid = threadIdx.x;
    const int wid = tid >> 5, lane = tid & 31;
    const int kh = lane >> 4, l16 = lane & 15;
    const int waveM = wid & 3, waveN = wid >> 2;
    const int m0 = blockIdx.y * 128;     // j
    const int n0 = blockIdx.x * 64;      // f

    f32x8 acc[2][2] = {};

    for (int kb = 0; kb < FEAT; kb += 32) {
        #pragma unroll
        for (int i = 0; i < 4; ++i) {
            int linear = (i * 256 + tid) * 4;
            int row = linear >> 5, col = linear & 31;
            int j = m0 + row;
            float4 v = {0.f, 0.f, 0.f, 0.f};
            if (j < NALL) v = *(const float4*)&protos[(size_t)j * FEAT + kb + col];
            split_store4(v, &sAhi[row * 40 + col], &sAlo[row * 40 + col]);
        }
        #pragma unroll
        for (int i = 0; i < 2; ++i) {
            int linear = (i * 256 + tid) * 4;
            int krow = linear >> 6, fcol = linear & 63;
            float4 v = *(const float4*)&encw[(size_t)(kb + krow) * FEAT + n0 + fcol];
            unsigned short h, l;
            split2(v.x, h, l); sBhi[(fcol + 0) * 40 + krow] = h; sBlo[(fcol + 0) * 40 + krow] = l;
            split2(v.y, h, l); sBhi[(fcol + 1) * 40 + krow] = h; sBlo[(fcol + 1) * 40 + krow] = l;
            split2(v.z, h, l); sBhi[(fcol + 2) * 40 + krow] = h; sBlo[(fcol + 2) * 40 + krow] = l;
            split2(v.w, h, l); sBhi[(fcol + 3) * 40 + krow] = h; sBlo[(fcol + 3) * 40 + krow] = l;
        }
        __syncthreads();

        Frag aH[2], aL[2], bH[2], bL[2];
        #pragma unroll
        for (int ms = 0; ms < 2; ++ms) {
            int r = waveM * 32 + ms * 16 + l16;
            load_frag_a(sAhi, r, kh, aH[ms]);
            load_frag_a(sAlo, r, kh, aL[ms]);
        }
        #pragma unroll
        for (int ns = 0; ns < 2; ++ns) {
            int c = waveN * 32 + ns * 16 + l16;
            load_frag_b(sBhi, c, kh, bH[ns]);
            load_frag_b(sBlo, c, kh, bL[ns]);
        }
        #pragma unroll
        for (int ms = 0; ms < 2; ++ms)
            #pragma unroll
            for (int ns = 0; ns < 2; ++ns) {
                acc[ms][ns] = wmma_bf16(aH[ms], bH[ns], acc[ms][ns]);
                acc[ms][ns] = wmma_bf16(aH[ms], bL[ns], acc[ms][ns]);
                acc[ms][ns] = wmma_bf16(aL[ms], bH[ns], acc[ms][ns]);
            }
        __syncthreads();
    }

    #pragma unroll
    for (int ms = 0; ms < 2; ++ms)
        #pragma unroll
        for (int ns = 0; ns < 2; ++ns) {
            int f = n0 + waveN * 32 + ns * 16 + l16;
            float b = encb[f];
            #pragma unroll
            for (int v = 0; v < 8; ++v) {
                int j = m0 + waveM * 32 + ms * 16 + kh * 8 + v;
                float val = acc[ms][ns][v] + b;
                unsigned short h, l;
                split2(val, h, l);
                ep_hi[(size_t)j * FEAT + f] = h;
                ep_lo[(size_t)j * FEAT + f] = l;
            }
        }
}

// ---------- kernel: main GEMM + distance epilogue ----------
// block: 256 thr (8 waves), tile 128(b) x 64(j), K step 32,
// double-buffered LDS filled by async global->LDS b128 copies (ASYNCcnt pipelined).
__global__ void __launch_bounds__(256) dist_gemm_kernel(const unsigned short* __restrict__ xhi,
                                                        const unsigned short* __restrict__ xlo,
                                                        const unsigned short* __restrict__ ep_hi,
                                                        const unsigned short* __restrict__ ep_lo,
                                                        const float* __restrict__ xsq,
                                                        const float* __restrict__ psq,
                                                        float* __restrict__ dmat) {
    __shared__ __align__(16) unsigned short sA[2][2][128 * 40];  // [buf][hi/lo]
    __shared__ __align__(16) unsigned short sB[2][2][64 * 40];

    const int tid = threadIdx.x;
    const int wid = tid >> 5, lane = tid & 31;
    const int kh = lane >> 4, l16 = lane & 15;
    const int waveM = wid & 3, waveN = wid >> 2;
    const int m0 = blockIdx.y * 128;     // batch rows
    const int n0 = blockIdx.x * 64;      // proto cols (padded 1024)

    // per-thread copy slots: A tile 128x32 ush per array -> 2 b128/thread; B tile 64x32 -> 1
    const int aRow = tid >> 2, aCol = (tid & 3) * 8;   // rows 0..63 (pass0), +64 (pass1)
    const int bRow = tid >> 2, bCol = (tid & 3) * 8;   // rows 0..63

    f32x8 acc[2][2] = {};

    auto issue_stage = [&](int buf, int kb) {
        const size_t gA0 = (size_t)(m0 + aRow) * FEAT + kb + aCol;
        const size_t gA1 = gA0 + (size_t)64 * FEAT;
        const size_t gB  = (size_t)(n0 + bRow) * FEAT + kb + bCol;
        async_copy_b128(&xhi[gA0],  &sA[buf][0][aRow * 40 + aCol]);
        async_copy_b128(&xhi[gA1],  &sA[buf][0][(aRow + 64) * 40 + aCol]);
        async_copy_b128(&xlo[gA0],  &sA[buf][1][aRow * 40 + aCol]);
        async_copy_b128(&xlo[gA1],  &sA[buf][1][(aRow + 64) * 40 + aCol]);
        async_copy_b128(&ep_hi[gB], &sB[buf][0][bRow * 40 + bCol]);
        async_copy_b128(&ep_lo[gB], &sB[buf][1][bRow * 40 + bCol]);
    };

    issue_stage(0, 0);
    int buf = 0;
    for (int kb = 0; kb < FEAT; kb += 32) {
        if (kb + 32 < FEAT) {
            issue_stage(buf ^ 1, kb + 32);   // prefetch next stage into other buffer
            WAIT_ASYNC(6);                   // retire current stage (6 async ops/wave ahead)
        } else {
            WAIT_ASYNC(0);
        }
        __syncthreads();

        Frag aH[2], aL[2], bH[2], bL[2];
        #pragma unroll
        for (int ms = 0; ms < 2; ++ms) {
            int r = waveM * 32 + ms * 16 + l16;
            load_frag_a(sA[buf][0], r, kh, aH[ms]);
            load_frag_a(sA[buf][1], r, kh, aL[ms]);
        }
        #pragma unroll
        for (int ns = 0; ns < 2; ++ns) {
            int c = waveN * 32 + ns * 16 + l16;
            load_frag_b(sB[buf][0], c, kh, bH[ns]);
            load_frag_b(sB[buf][1], c, kh, bL[ns]);
        }
        #pragma unroll
        for (int ms = 0; ms < 2; ++ms)
            #pragma unroll
            for (int ns = 0; ns < 2; ++ns) {
                acc[ms][ns] = wmma_bf16(aH[ms], bH[ns], acc[ms][ns]);
                acc[ms][ns] = wmma_bf16(aH[ms], bL[ns], acc[ms][ns]);
                acc[ms][ns] = wmma_bf16(aL[ms], bH[ns], acc[ms][ns]);
            }
        __syncthreads();   // all waves done reading buf before it is refilled
        buf ^= 1;
    }

    // epilogue: d = sqrt(max(xsq - 2*s + psq, 0))
    #pragma unroll
    for (int ms = 0; ms < 2; ++ms)
        #pragma unroll
        for (int ns = 0; ns < 2; ++ns) {
            int n = n0 + waveN * 32 + ns * 16 + l16;
            float ps = psq[n];
            #pragma unroll
            for (int v = 0; v < 8; ++v) {
                int m = m0 + waveM * 32 + ms * 16 + kh * 8 + v;
                float s = acc[ms][ns][v];
                float d2 = xsq[m] - 2.0f * s + ps;
                dmat[(size_t)m * NPAD + n] = sqrtf(fmaxf(d2, 0.0f));
            }
        }
}

// ---------- kernel: min over 10 prototypes ----------
__global__ void __launch_bounds__(256) min_reduce_kernel(const float* __restrict__ dmat,
                                                         float* __restrict__ out) {
    int gid = blockIdx.x * blockDim.x + threadIdx.x;
    if (gid >= BATCH * NCLS) return;
    int b = gid / NCLS;
    int c = gid - b * NCLS;
    const float* row = dmat + (size_t)b * NPAD;
    float m = row[c];
    #pragma unroll
    for (int p = 1; p < NPROT; ++p) m = fminf(m, row[p * NCLS + c]);
    out[gid] = m;
}

extern "C" void kernel_launch(void* const* d_in, const int* in_sizes, int n_in,
                              void* d_out, int out_size, void* d_ws, size_t ws_size,
                              hipStream_t stream) {
    const float* x      = (const float*)d_in[0];
    const float* protos = (const float*)d_in[1];
    const float* enc_w  = (const float*)d_in[2];
    const float* enc_b  = (const float*)d_in[3];
    float* out = (float*)d_out;

    char* ws = (char*)d_ws;
    unsigned short* ep_hi = (unsigned short*)(ws);                                   // 2 MB
    unsigned short* ep_lo = (unsigned short*)(ws + (size_t)2 * 1024 * 1024);         // 2 MB
    float* psq  = (float*)(ws + (size_t)4 * 1024 * 1024);                            // 4 KB
    float* xsq  = (float*)(ws + (size_t)4 * 1024 * 1024 + 16384);                    // 64 KB
    unsigned short* x_hi = (unsigned short*)(ws + (size_t)8 * 1024 * 1024);          // 32 MB
    unsigned short* x_lo = (unsigned short*)(ws + (size_t)40 * 1024 * 1024);         // 32 MB
    float* dmat = (float*)(ws + (size_t)72 * 1024 * 1024);                           // 64 MB

    // 1) split x -> bf16 hi/lo + ||x||^2 per row (single pass over x)
    xsplit_kernel<<<BATCH / 8, 256, 0, stream>>>(x, x_hi, x_lo, xsq);
    // 2) ep = protos @ enc_w + enc_b, split into bf16 hi/lo (padded to 1024 rows)
    enc_gemm_kernel<<<dim3(FEAT / 64, NPAD / 128), 256, 0, stream>>>(protos, enc_w, enc_b,
                                                                     ep_hi, ep_lo);
    // 3) ||ep||^2 per row
    psq_kernel<<<NPAD / 8, 256, 0, stream>>>(ep_hi, ep_lo, psq);
    // 4) main GEMM (bf16x3) + distance epilogue -> dmat (B x 1024 padded)
    dist_gemm_kernel<<<dim3(NPAD / 64, BATCH / 128), 256, 0, stream>>>(x_hi, x_lo, ep_hi, ep_lo,
                                                                       xsq, psq, dmat);
    // 5) min over 10 prototypes -> (B, 100)
    min_reduce_kernel<<<(BATCH * NCLS + 255) / 256, 256, 0, stream>>>(dmat, out);
}
